// EncoderLayer_59725815218773
// MI455X (gfx1250) — matrix operand order repaired and checked by reading, not compiled, not run on hardware
//
#include <hip/hip_runtime.h>

typedef __attribute__((ext_vector_type(16))) __bf16 v16bf;
typedef __attribute__((ext_vector_type(8)))  __bf16 v8bf;
typedef __attribute__((ext_vector_type(8)))  float  v8f;

static constexpr int kDM    = 1024;
static constexpr int kHEADS = 16;
static constexpr int kDK    = 64;
static constexpr int kDFF   = 4096;

// ---------------------------------------------------------------- helpers ---

__device__ __forceinline__ unsigned short f2bf(float f) {
  unsigned int u = __float_as_uint(f);
  u += 0x7FFFu + ((u >> 16) & 1u);   // round-to-nearest-even
  return (unsigned short)(u >> 16);
}

__device__ __forceinline__ v8f wmma_bf(v16bf a, v16bf b, v8f c) {
  return __builtin_amdgcn_wmma_f32_16x16x32_bf16(
      false, a, false, b, (short)0, c, false, false);
}

// XOR-butterfly within a 16-lane DPP row: single VALU op (ROW_XMASK).
template <int MASK>
__device__ __forceinline__ float dpp_xor_f(float x) {
  int i = __builtin_amdgcn_update_dpp(0, __float_as_int(x),
                                      0x160 | MASK, 0xf, 0xf, true);
  return __int_as_float(i);
}

union frag16 { v16bf v; v8bf h[2]; };

// A-fragment: 16x32 bf16 tile, row-major (ld in ushorts). Lanes 0-15 hold
// K=0..7 & 16..23; lanes 16-31 hold K=8..15 & 24..31.
__device__ __forceinline__ v16bf ld_frag_a(const unsigned short* base, int row,
                                           int ld, int koff) {
  int lane = threadIdx.x & 31;
  int r = lane & 15, hf = lane >> 4;
  const v8bf* p = (const v8bf*)(base + (long)(row + r) * ld + koff + hf * 8);
  frag16 u;
  u.h[0] = p[0];
  u.h[1] = p[2];   // +16 ushorts
  return u.v;
}

// B-fragment: 32x16 bf16 tile stored N-major (Bt[n][k]). Lanes 0-15 hold
// K=0..7 & 16..23; lanes 16-31 hold K=8..15 & 24..31, column n = lane&15.
__device__ __forceinline__ v16bf ld_frag_b(const unsigned short* base, int row,
                                           int ld, int koff) {
  int lane = threadIdx.x & 31;
  int r = lane & 15, hf = lane >> 4;
  const v8bf* p = (const v8bf*)(base + (long)(row + r) * ld + koff + hf * 8);
  frag16 u;
  u.h[0] = p[0];
  u.h[1] = p[2];
  return u.v;
}

// ----------------------------------------------------------- fp32 -> bf16 ---

__global__ void cvt_bf16_kernel(const float* __restrict__ in,
                                unsigned short* __restrict__ out, long n) {
  long i = (long)blockIdx.x * blockDim.x + threadIdx.x;
  if (i < n) out[i] = f2bf(in[i]);
}

// Transposing convert: W[K][N] fp32 -> Wt[N][K] bf16 (one-time weight prep).
__global__ void cvt_bf16_t_kernel(const float* __restrict__ in,
                                  unsigned short* __restrict__ out,
                                  int Kd, int Nd) {
  long i = (long)blockIdx.x * blockDim.x + threadIdx.x;
  long n = (long)Kd * Nd;
  if (i < n) {
    int k = (int)(i / Nd), c = (int)(i % Nd);
    out[(long)c * Kd + k] = f2bf(in[i]);
  }
}

// ------------------------------------------------------------------- GEMM ---
// C[M,N] = A[M,K](bf16 row-major) * Wt[N,K](bf16 pre-transposed) + bias.
// 128x128 block tile, K-tile 32, 8 waves; double-buffered LDS so global loads
// for tile t+1 overlap the WMMAs on tile t (one barrier per K-tile).

template <bool RELU, bool OUTF, bool OUTB, bool TOUT>
__global__ void __launch_bounds__(256)
gemm_bf16_kernel(const unsigned short* __restrict__ A,
                 const unsigned short* __restrict__ Bt_g,
                 const float* __restrict__ bias,
                 float* __restrict__ outF,
                 unsigned short* __restrict__ outB,
                 int M, int N, int K) {
  __shared__ alignas(16) unsigned short As[2][128][40];
  __shared__ alignas(16) unsigned short Bs[2][128][40];

  const int tid  = threadIdx.x;
  const int wid  = tid >> 5;
  const int lane = tid & 31;
  const int bm = blockIdx.y * 128;
  const int bn = blockIdx.x * 128;
  const int wm = (wid & 1) * 64;
  const int wn = (wid >> 1) * 32;

  v8f acc[4][2];
#pragma unroll
  for (int i = 0; i < 4; i++)
#pragma unroll
    for (int j = 0; j < 2; j++)
#pragma unroll
      for (int v = 0; v < 8; v++) acc[i][j][v] = 0.f;

  const int r2 = tid >> 1, c2 = (tid & 1) * 16;  // both tiles: 128 rows x 32 k

  const unsigned short* gA = A    + (long)(bm + r2) * K + c2;
  const unsigned short* gB = Bt_g + (long)(bn + r2) * K + c2;

  // preload tile 0
  {
    const uint4* ga = (const uint4*)gA;
    const uint4* gb = (const uint4*)gB;
    uint4 a0 = ga[0], a1 = ga[1], b0 = gb[0], b1 = gb[1];
    *(uint4*)&As[0][r2][c2]     = a0;
    *(uint4*)&As[0][r2][c2 + 8] = a1;
    *(uint4*)&Bs[0][r2][c2]     = b0;
    *(uint4*)&Bs[0][r2][c2 + 8] = b1;
  }
  __syncthreads();

  const int nt = K >> 5;
  for (int t = 0; t < nt; t++) {
    const int cur = t & 1, nxt = cur ^ 1;
    uint4 a0n, a1n, b0n, b1n;
    const bool more = (t + 1) < nt;
    if (more) {
      const uint4* ga = (const uint4*)(gA + (t + 1) * 32);
      const uint4* gb = (const uint4*)(gB + (t + 1) * 32);
      a0n = ga[0]; a1n = ga[1]; b0n = gb[0]; b1n = gb[1];
      if ((t + 2) < nt) {
        __builtin_prefetch((const void*)(gA + (t + 2) * 32), 0, 1);
        __builtin_prefetch((const void*)(gB + (t + 2) * 32), 0, 1);
      }
    }

    v16bf af[4], bf[2];
#pragma unroll
    for (int mi = 0; mi < 4; mi++)
      af[mi] = ld_frag_a(&As[cur][0][0], wm + mi * 16, 40, 0);
#pragma unroll
    for (int ni = 0; ni < 2; ni++)
      bf[ni] = ld_frag_b(&Bs[cur][0][0], wn + ni * 16, 40, 0);
#pragma unroll
    for (int mi = 0; mi < 4; mi++)
#pragma unroll
      for (int ni = 0; ni < 2; ni++)
        acc[mi][ni] = wmma_bf(af[mi], bf[ni], acc[mi][ni]);

    if (more) {
      *(uint4*)&As[nxt][r2][c2]     = a0n;
      *(uint4*)&As[nxt][r2][c2 + 8] = a1n;
      *(uint4*)&Bs[nxt][r2][c2]     = b0n;
      *(uint4*)&Bs[nxt][r2][c2 + 8] = b1n;
    }
    __syncthreads();
  }

  // Epilogue: C layout -> row m = v + 8*(lane>>4), col n = lane&15
  const int cn = lane & 15, hv = lane >> 4;
#pragma unroll
  for (int mi = 0; mi < 4; mi++) {
#pragma unroll
    for (int ni = 0; ni < 2; ni++) {
      int col = bn + wn + ni * 16 + cn;
      float bv = bias[col];
#pragma unroll
      for (int v = 0; v < 8; v++) {
        int row = bm + wm + mi * 16 + v + 8 * hv;
        float val = acc[mi][ni][v] + bv;
        if (RELU) val = fmaxf(val, 0.f);
        if (OUTF) outF[(long)row * N + col] = val;
        if (OUTB) {
          if (TOUT) outB[(long)col * M + row] = f2bf(val);
          else      outB[(long)row * N + col] = f2bf(val);
        }
      }
    }
  }
}

// -------------------------------------------------------- flash attention ---
// One block = 4 waves = 64 queries for one (b,h). Streams 32-key tiles with
// double-buffered K/V LDS, online softmax (DPP row-xmask reductions), WMMA
// for Q*K^T and P*V. Q/K bf16 [B,L,DM]; V pre-transposed Vt[DM][B*L].

__global__ void __launch_bounds__(128)
attn_kernel(const unsigned short* __restrict__ Qb,
            const unsigned short* __restrict__ Kb,
            const unsigned short* __restrict__ VtG,
            unsigned short* __restrict__ ctxB, int L, int R) {
  __shared__ alignas(16) unsigned short Ks[2][32][72];     // [key][d]
  __shared__ alignas(16) unsigned short Vt[2][64][40];     // [d][key]
  __shared__ alignas(16) unsigned short Ps[4][16][40];     // per-wave probs

  const int tid = threadIdx.x, wid = tid >> 5, lane = tid & 31;
  const int b = blockIdx.y / kHEADS, h = blockIdx.y % kHEADS;
  const int q0 = blockIdx.x * 64 + wid * 16;
  const int r = lane & 15, hf = lane >> 4;

  // Q fragments straight from global (A-fragment access pattern)
  v16bf qf[2];
  {
    const unsigned short* qrow = Qb + ((long)(b * L + q0 + r)) * kDM + h * kDK;
#pragma unroll
    for (int kb = 0; kb < 2; kb++) {
      const v8bf* p = (const v8bf*)(qrow + kb * 32 + hf * 8);
      frag16 u; u.h[0] = p[0]; u.h[1] = p[2];
      qf[kb] = u.v;
    }
  }

  v8f oacc[4];
  float mrow[8], lrow[8];
#pragma unroll
  for (int dt = 0; dt < 4; dt++)
#pragma unroll
    for (int v = 0; v < 8; v++) oacc[dt][v] = 0.f;
#pragma unroll
  for (int v = 0; v < 8; v++) { mrow[v] = -1e30f; lrow[v] = 0.f; }

  const int kr = tid >> 2, kc = (tid & 3) * 16;   // K tile: 32 keys x 64 d
  const int vd = tid >> 1, vk = (tid & 1) * 16;   // V tile: 64 d x 32 keys

  const unsigned short* gK = Kb  + ((long)(b * L + kr)) * kDM + h * kDK + kc;
  const unsigned short* gV = VtG + ((long)(h * kDK + vd)) * R + b * L + vk;

  // preload tile 0
  {
    const uint4* gk = (const uint4*)gK;
    const uint4* gv = (const uint4*)gV;
    uint4 k0 = gk[0], k1 = gk[1], v0 = gv[0], v1 = gv[1];
    *(uint4*)&Ks[0][kr][kc]     = k0;
    *(uint4*)&Ks[0][kr][kc + 8] = k1;
    *(uint4*)&Vt[0][vd][vk]     = v0;
    *(uint4*)&Vt[0][vd][vk + 8] = v1;
  }
  __syncthreads();

  const int nt = L >> 5;
  for (int t = 0; t < nt; t++) {
    const int cur = t & 1, nxt = cur ^ 1;
    uint4 k0n, k1n, v0n, v1n;
    const bool more = (t + 1) < nt;
    if (more) {
      const uint4* gk = (const uint4*)(gK + (long)(t + 1) * 32 * kDM);
      const uint4* gv = (const uint4*)(gV + (t + 1) * 32);
      k0n = gk[0]; k1n = gk[1]; v0n = gv[0]; v1n = gv[1];
      if ((t + 2) < nt) {
        __builtin_prefetch((const void*)(gK + (long)(t + 2) * 32 * kDM), 0, 1);
        __builtin_prefetch((const void*)(gV + (t + 2) * 32), 0, 1);
      }
    }

    // S = Q(16x64) * K^T : K rows serve directly as B fragments (n = key)
    v8f s0, s1;
    {
      v8f z;
#pragma unroll
      for (int v = 0; v < 8; v++) z[v] = 0.f;
      s0 = wmma_bf(qf[0], ld_frag_b(&Ks[cur][0][0], 0, 72, 0), z);
      s0 = wmma_bf(qf[1], ld_frag_b(&Ks[cur][0][0], 0, 72, 32), s0);
      s1 = wmma_bf(qf[0], ld_frag_b(&Ks[cur][0][0], 16, 72, 0), z);
      s1 = wmma_bf(qf[1], ld_frag_b(&Ks[cur][0][0], 16, 72, 32), s1);
    }

    // Online softmax over the 32-key slab; row reductions = DPP row_xmask
    float corr[8];
#pragma unroll
    for (int v = 0; v < 8; v++) {
      float a0 = s0[v] * 0.125f;                 // 1/sqrt(64)
      float a1 = s1[v] * 0.125f;
      float mx = fmaxf(a0, a1);
      mx = fmaxf(mx, dpp_xor_f<1>(mx));
      mx = fmaxf(mx, dpp_xor_f<2>(mx));
      mx = fmaxf(mx, dpp_xor_f<4>(mx));
      mx = fmaxf(mx, dpp_xor_f<8>(mx));
      float nm = fmaxf(mrow[v], mx);
      float p0 = __expf(a0 - nm);
      float p1 = __expf(a1 - nm);
      float c  = __expf(mrow[v] - nm);
      float ps = p0 + p1;
      ps += dpp_xor_f<1>(ps);
      ps += dpp_xor_f<2>(ps);
      ps += dpp_xor_f<4>(ps);
      ps += dpp_xor_f<8>(ps);
      lrow[v] = lrow[v] * c + ps;
      mrow[v] = nm;
      corr[v] = c;
      Ps[wid][v + 8 * hf][r]      = f2bf(p0);    // C-layout -> row-major LDS
      Ps[wid][v + 8 * hf][16 + r] = f2bf(p1);
    }
#pragma unroll
    for (int dt = 0; dt < 4; dt++)
#pragma unroll
      for (int v = 0; v < 8; v++) oacc[dt][v] *= corr[v];

    // Wave-local LDS RAW (DS ops in-order per wave; explicit wait anyway)
    asm volatile("s_wait_dscnt 0" ::: "memory");

    // O += P(16x32) * V(32x64)
    v16bf pf = ld_frag_a(&Ps[wid][0][0], 0, 40, 0);
#pragma unroll
    for (int dt = 0; dt < 4; dt++) {
      v16bf vf = ld_frag_b(&Vt[cur][0][0], dt * 16, 40, 0);
      oacc[dt] = wmma_bf(pf, vf, oacc[dt]);
    }

    if (more) {
      *(uint4*)&Ks[nxt][kr][kc]     = k0n;
      *(uint4*)&Ks[nxt][kr][kc + 8] = k1n;
      *(uint4*)&Vt[nxt][vd][vk]     = v0n;
      *(uint4*)&Vt[nxt][vd][vk + 8] = v1n;
    }
    __syncthreads();
  }

  // Normalize and emit ctx (bf16 feeds the O-projection GEMM)
#pragma unroll
  for (int v = 0; v < 8; v++) {
    float inv = 1.0f / lrow[v];
    int q = q0 + v + 8 * hf;
    unsigned short* orow = ctxB + ((long)(b * L + q)) * kDM + h * kDK;
#pragma unroll
    for (int dt = 0; dt < 4; dt++)
      orow[dt * 16 + r] = f2bf(oacc[dt][v] * inv);
  }
}

// --------------------------------------------------- residual + layernorm ---

template <bool OUTB>
__global__ void __launch_bounds__(256)
add_ln_kernel(const float* __restrict__ x, const float* __restrict__ y,
              const float* __restrict__ gam, const float* __restrict__ bet,
              float* __restrict__ outF, unsigned short* __restrict__ outB) {
  __shared__ float rs[8], rs2[8];
  const long row = blockIdx.x;
  const float* xr = x + row * kDM;
  const float* yr = y + row * kDM;
  float vals[4];
  float s = 0.f, s2 = 0.f;
#pragma unroll
  for (int i = 0; i < 4; i++) {
    int c = threadIdx.x + i * 256;
    float v = xr[c] + yr[c];
    vals[i] = v; s += v; s2 += v * v;
  }
  // in-wave reduction: DPP for masks 1..8, shuffle for the cross-half step
  s += dpp_xor_f<1>(s);  s2 += dpp_xor_f<1>(s2);
  s += dpp_xor_f<2>(s);  s2 += dpp_xor_f<2>(s2);
  s += dpp_xor_f<4>(s);  s2 += dpp_xor_f<4>(s2);
  s += dpp_xor_f<8>(s);  s2 += dpp_xor_f<8>(s2);
  s += __shfl_xor(s, 16, 32);
  s2 += __shfl_xor(s2, 16, 32);
  if ((threadIdx.x & 31) == 0) { rs[threadIdx.x >> 5] = s; rs2[threadIdx.x >> 5] = s2; }
  __syncthreads();
  float ts = 0.f, ts2 = 0.f;
#pragma unroll
  for (int i = 0; i < 8; i++) { ts += rs[i]; ts2 += rs2[i]; }
  float mu  = ts * (1.0f / kDM);
  float var = ts2 * (1.0f / kDM) - mu * mu;
  float inv = rsqrtf(var + 1e-5f);
#pragma unroll
  for (int i = 0; i < 4; i++) {
    int c = threadIdx.x + i * 256;
    float v = (vals[i] - mu) * inv * gam[c] + bet[c];
    long idx = row * kDM + c;
    outF[idx] = v;
    if (OUTB) outB[idx] = f2bf(v);
  }
}

// --------------------------------------------------------------- launcher ---

extern "C" void kernel_launch(void* const* d_in, const int* in_sizes, int n_in,
                              void* d_out, int out_size, void* d_ws, size_t ws_size,
                              hipStream_t stream) {
  (void)in_sizes; (void)n_in; (void)out_size; (void)ws_size;

  const float* src = (const float*)d_in[0];
  const float* Wq  = (const float*)d_in[1];
  const float* bq  = (const float*)d_in[2];
  const float* Wk  = (const float*)d_in[3];
  const float* bk  = (const float*)d_in[4];
  const float* Wv  = (const float*)d_in[5];
  const float* bv  = (const float*)d_in[6];
  const float* Wo  = (const float*)d_in[7];
  const float* bo  = (const float*)d_in[8];
  const float* W1  = (const float*)d_in[9];
  const float* b1  = (const float*)d_in[10];
  const float* W2  = (const float*)d_in[11];
  const float* b2  = (const float*)d_in[12];
  const float* g1  = (const float*)d_in[13];
  const float* be1 = (const float*)d_in[14];
  const float* g2  = (const float*)d_in[15];
  const float* be2 = (const float*)d_in[16];

  const int Bb = 4, L = 2048;
  const long R = (long)Bb * L;  // 8192 token rows

  char* w = (char*)d_ws;
  auto take = [&](size_t bytes) -> void* {
    void* p = (void*)w;
    w += (bytes + 255) & ~(size_t)255;
    return p;
  };
  unsigned short* srcb = (unsigned short*)take(R * kDM * 2);
  unsigned short* Wqt  = (unsigned short*)take((size_t)kDM * kDM * 2);   // [N][K]
  unsigned short* Wkt  = (unsigned short*)take((size_t)kDM * kDM * 2);
  unsigned short* Wvt  = (unsigned short*)take((size_t)kDM * kDM * 2);
  unsigned short* Wot  = (unsigned short*)take((size_t)kDM * kDM * 2);
  unsigned short* W1t  = (unsigned short*)take((size_t)kDM * kDFF * 2);  // [DFF][DM]
  unsigned short* W2t  = (unsigned short*)take((size_t)kDFF * kDM * 2);  // [DM][DFF]
  unsigned short* Qbf  = (unsigned short*)take(R * kDM * 2);
  unsigned short* Kbf  = (unsigned short*)take(R * kDM * 2);
  unsigned short* Vtb  = (unsigned short*)take(R * kDM * 2);             // [DM][R]
  unsigned short* ctxb = (unsigned short*)take(R * kDM * 2);
  float*          ao   = (float*)take(R * kDM * 4);
  float*          xf   = (float*)take(R * kDM * 4);
  unsigned short* xb   = (unsigned short*)take(R * kDM * 2);
  unsigned short* hb   = (unsigned short*)take(R * kDFF * 2);
  float*          ff   = (float*)take(R * kDM * 4);

  cvt_bf16_kernel<<<dim3((unsigned)((R * kDM + 255) / 256)), dim3(256), 0, stream>>>(
      src, srcb, R * kDM);
  auto cvt_t = [&](const float* in, unsigned short* out, int Kd, int Nd) {
    long n = (long)Kd * Nd;
    cvt_bf16_t_kernel<<<dim3((unsigned)((n + 255) / 256)), dim3(256), 0, stream>>>(
        in, out, Kd, Nd);
  };
  cvt_t(Wq, Wqt, kDM, kDM);
  cvt_t(Wk, Wkt, kDM, kDM);
  cvt_t(Wv, Wvt, kDM, kDM);
  cvt_t(Wo, Wot, kDM, kDM);
  cvt_t(W1, W1t, kDM, kDFF);
  cvt_t(W2, W2t, kDFF, kDM);

  // QKV projections (bf16 out; V written pre-transposed)
  gemm_bf16_kernel<false, false, true, false>
      <<<dim3(kDM / 128, (unsigned)(R / 128)), dim3(256), 0, stream>>>(
          srcb, Wqt, bq, nullptr, Qbf, (int)R, kDM, kDM);
  gemm_bf16_kernel<false, false, true, false>
      <<<dim3(kDM / 128, (unsigned)(R / 128)), dim3(256), 0, stream>>>(
          srcb, Wkt, bk, nullptr, Kbf, (int)R, kDM, kDM);
  gemm_bf16_kernel<false, false, true, true>
      <<<dim3(kDM / 128, (unsigned)(R / 128)), dim3(256), 0, stream>>>(
          srcb, Wvt, bv, nullptr, Vtb, (int)R, kDM, kDM);

  // Flash attention -> ctx bf16
  attn_kernel<<<dim3(L / 64, Bb * kHEADS), dim3(128), 0, stream>>>(
      Qbf, Kbf, Vtb, ctxb, L, (int)R);

  // O projection -> fp32
  gemm_bf16_kernel<false, true, false, false>
      <<<dim3(kDM / 128, (unsigned)(R / 128)), dim3(256), 0, stream>>>(
          ctxb, Wot, bo, ao, nullptr, (int)R, kDM, kDM);

  // x = LN(src + attn_out), fp32 + bf16
  add_ln_kernel<true><<<dim3((unsigned)R), dim3(256), 0, stream>>>(
      src, ao, g1, be1, xf, xb);

  // FFN
  gemm_bf16_kernel<true, false, true, false>
      <<<dim3(kDFF / 128, (unsigned)(R / 128)), dim3(256), 0, stream>>>(
          xb, W1t, b1, nullptr, hb, (int)R, kDFF, kDM);
  gemm_bf16_kernel<false, true, false, false>
      <<<dim3(kDM / 128, (unsigned)(R / 128)), dim3(256), 0, stream>>>(
          hb, W2t, b2, ff, nullptr, (int)R, kDM, kDFF);

  // out = LN(x + ff)
  add_ln_kernel<false><<<dim3((unsigned)R), dim3(256), 0, stream>>>(
      xf, ff, g2, be2, (float*)d_out, nullptr);
}
